// MambaIR_Block_37495064494609
// MI455X (gfx1250) — matrix operand order repaired and checked by reading, not compiled, not run on hardware
//
#include <hip/hip_runtime.h>

// ---------------------------------------------------------------------------
// MambaIR block, CDNA5 (gfx1250, wave32).
//   K0 : weight transpose/pad -> bf16 (column-major so B-fragments are
//        K-contiguous), A = -exp(A_log)
//   K0b: x (NCHW f32) -> token-major bf16 (A-fragments become 2x b128 loads)
//   K1 : fused xp@W_in -> LayerNorm -> u@W_dt -> softplus(dt@W_dtr)
//        (v_wmma_f32_16x16x32_bf16, one wave per 16-token tile)
//   K2 : depthwise 3x3 conv + z*silu(z), token-major (coalesced)
//   K3a/b/c: chunked selective scan; K3c emits ybf = bf16(y + u*D + xc)
//   K4 : ybf@W_out + x residual, transposed store
// ---------------------------------------------------------------------------

typedef __attribute__((ext_vector_type(16))) __bf16 v16bf;
typedef __attribute__((ext_vector_type(8)))  __bf16 v8bf;
typedef __attribute__((ext_vector_type(8)))  float  v8f;

#define C_   128
#define N_   64
#define L_   4096
#define NCH  64
#define CHL  64
#define EPS_ 1e-5f

__device__ __forceinline__ v16bf cat16(v8bf lo, v8bf hi) {
  return __builtin_shufflevector(lo, hi, 0,1,2,3,4,5,6,7,8,9,10,11,12,13,14,15);
}
// A-matrix bf16 16x32 fragment: row-contiguous activation memory.
// element j (0..15) <-> K = kt*32 + 16*half + j
__device__ __forceinline__ v16bf lda16(const __bf16* row, int kt, int half) {
  v8bf lo = *(const v8bf*)(row + kt * 32 + half * 16);
  v8bf hi = *(const v8bf*)(row + kt * 32 + half * 16 + 8);
  return cat16(lo, hi);
}
// B-matrix bf16 32x16 fragment from column-major (transposed) weights:
// element j <-> K = kt*32 + 16*half + j, all 16 contiguous.
__device__ __forceinline__ v16bf ldb16(const __bf16* p) {
  v8bf lo = *(const v8bf*)p;
  v8bf hi = *(const v8bf*)(p + 8);
  return cat16(lo, hi);
}
__device__ __forceinline__ v8f wmma_bf16(v16bf a, v16bf b, v8f c) {
  return __builtin_amdgcn_wmma_f32_16x16x32_bf16(false, a, false, b,
                                                 (short)0, c, false, false);
}

// ------------------------------------------- K0: weights -> bf16 transposed
__global__ void k0_prep(const float* __restrict__ W_in, const float* __restrict__ W_dt,
                        const float* __restrict__ W_dtr, const float* __restrict__ W_out,
                        const float* __restrict__ A_log,
                        __bf16* wWinT, __bf16* wWdtT, __bf16* wWdtrT, __bf16* wWoutT,
                        float* Aneg)
{
  int t = blockIdx.x * blockDim.x + threadIdx.x;
  if (t < 32768) {                            // W_in^T : [256][128]
    int n = t >> 7, k = t & 127;
    wWinT[t] = (__bf16)W_in[k * 256 + n];
    return;
  }
  t -= 32768;
  if (t < 18432) {                            // W_dt^T : [144][128], pad n>=136
    int n = t / 128, k = t % 128;
    wWdtT[t] = (n < 136) ? (__bf16)W_dt[k * 136 + n] : (__bf16)0.0f;
    return;
  }
  t -= 18432;
  if (t < 4096) {                             // W_dtr^T: [128][32], pad k>=8
    int n = t >> 5, k = t & 31;
    wWdtrT[t] = (k < 8) ? (__bf16)W_dtr[k * 128 + n] : (__bf16)0.0f;
    return;
  }
  t -= 4096;
  if (t < 16384) {                            // W_out^T: [128][128]
    int n = t >> 7, k = t & 127;
    wWoutT[t] = (__bf16)W_out[k * 128 + n];
    return;
  }
  t -= 16384;
  if (t < 8192) Aneg[t] = -expf(A_log[t]);
}

// -------------------------------- K0b: x (NCHW f32) -> token-major bf16
__global__ __launch_bounds__(256)
void k0b_xbf(const float* __restrict__ x, __bf16* __restrict__ xbf)
{
  const int t  = blockIdx.x * blockDim.x + threadIdx.x;   // linear over x
  const int hw = t & 4095, c = (t >> 12) & 127, b = t >> 19;
  xbf[(size_t)((b << 12) + hw) * C_ + c] = (__bf16)x[t];
}

// ------------------------------------------------- K1: fused projections
__global__ __launch_bounds__(128)
void k1_proj(const __bf16* __restrict__ xbf,
             const float* __restrict__ gamma, const float* __restrict__ beta,
             const float* __restrict__ b_dt,  const float* __restrict__ b_dtr,
             const __bf16* __restrict__ wWinT, const __bf16* __restrict__ wWdtT,
             const __bf16* __restrict__ wWdtrT,
             float* __restrict__ u_out, float* __restrict__ delta_out,
             float* __restrict__ Bs, float* __restrict__ Cs,
             float* __restrict__ xcv)
{
  __shared__ alignas(16) float  lds_u[4][16][C_];
  __shared__ alignas(16) __bf16 lds_ubf[4][16][C_];
  __shared__ alignas(16) __bf16 lds_dtraw[4][16][32];   // K padded 8->32

  // zero-fill dtraw pad (covered by the two __syncthreads before its use)
  {
    __bf16* p = &lds_dtraw[0][0][0] + threadIdx.x * 16;
    *(v8bf*)p = {}; *(v8bf*)(p + 8) = {};
  }

  const int lane = threadIdx.x & 31;
  const int wv   = threadIdx.x >> 5;
  const int t0   = (blockIdx.x * 4 + wv) * 16;          // 16-token tile
  const int m16  = lane & 15, half = lane >> 4;

  // ---- GEMM1: xp @ W_in  (A: 2x b128 per K-step, B: 2x b128)
  const __bf16* xrow = xbf + (size_t)(t0 + m16) * C_;
  v16bf afrag[4];
#pragma unroll
  for (int kt = 0; kt < 4; ++kt) afrag[kt] = lda16(xrow, kt, half);

  for (int nt = 0; nt < 16; ++nt) {
    const int col = nt * 16 + m16;
    const __bf16* bp = wWinT + col * 128 + half * 16;
    v8f acc = {};
#pragma unroll
    for (int kt = 0; kt < 4; ++kt)
      acc = wmma_bf16(afrag[kt], ldb16(bp + kt * 32), acc);
    if (nt < 8) {
#pragma unroll
      for (int i = 0; i < 8; ++i) lds_u[wv][i + half * 8][col] = acc[i];
    } else {
      const int c = col - C_;
#pragma unroll
      for (int i = 0; i < 8; ++i)
        xcv[(size_t)(t0 + i + half * 8) * C_ + c] = acc[i];   // token-major
    }
  }
  __syncthreads();

  // ---- LayerNorm over C (2 lanes per token, vectorized)
  {
    const int tk = lane & 15, sub = lane >> 4;
    const float* rowu = &lds_u[wv][tk][0];
    float s = 0.f, sq = 0.f;
    for (int c = sub * 64; c < sub * 64 + 64; c += 4) {
      float4 v = *(const float4*)(rowu + c);
      s  += v.x + v.y + v.z + v.w;
      sq += v.x * v.x + v.y * v.y + v.z * v.z + v.w * v.w;
    }
    s  += __shfl_xor(s, 16, 32);
    sq += __shfl_xor(sq, 16, 32);
    const float mu = s * (1.0f / C_);
    const float rs = rsqrtf(sq * (1.0f / C_) - mu * mu + EPS_);
    const int tok = t0 + tk;
    for (int c = sub * 64; c < sub * 64 + 64; c += 8) {
      float4 v0 = *(const float4*)(rowu + c);
      float4 v1 = *(const float4*)(rowu + c + 4);
      float4 g0 = *(const float4*)(gamma + c), g1 = *(const float4*)(gamma + c + 4);
      float4 e0 = *(const float4*)(beta + c),  e1 = *(const float4*)(beta + c + 4);
      float4 w0, w1;
      w0.x = (v0.x - mu) * rs * g0.x + e0.x;  w0.y = (v0.y - mu) * rs * g0.y + e0.y;
      w0.z = (v0.z - mu) * rs * g0.z + e0.z;  w0.w = (v0.w - mu) * rs * g0.w + e0.w;
      w1.x = (v1.x - mu) * rs * g1.x + e1.x;  w1.y = (v1.y - mu) * rs * g1.y + e1.y;
      w1.z = (v1.z - mu) * rs * g1.z + e1.z;  w1.w = (v1.w - mu) * rs * g1.w + e1.w;
      *(float4*)(u_out + (size_t)tok * C_ + c)     = w0;
      *(float4*)(u_out + (size_t)tok * C_ + c + 4) = w1;
      v8bf pk;
      pk[0] = (__bf16)w0.x; pk[1] = (__bf16)w0.y; pk[2] = (__bf16)w0.z; pk[3] = (__bf16)w0.w;
      pk[4] = (__bf16)w1.x; pk[5] = (__bf16)w1.y; pk[6] = (__bf16)w1.z; pk[7] = (__bf16)w1.w;
      *(v8bf*)&lds_ubf[wv][tk][c] = pk;
    }
  }
  __syncthreads();

  // ---- GEMM2: u @ W_dt + b_dt (N padded 136 -> 144)
  const __bf16* urow = &lds_ubf[wv][m16][0];
  v16bf a2[4];
#pragma unroll
  for (int kt = 0; kt < 4; ++kt) a2[kt] = lda16(urow, kt, half);

  for (int nt = 0; nt < 9; ++nt) {
    const int col = nt * 16 + m16;
    const float bias = (col < 136) ? b_dt[col] : 0.0f;
    v8f acc;
#pragma unroll
    for (int i = 0; i < 8; ++i) acc[i] = bias;
    const __bf16* bp = wWdtT + col * 128 + half * 16;
#pragma unroll
    for (int kt = 0; kt < 4; ++kt)
      acc = wmma_bf16(a2[kt], ldb16(bp + kt * 32), acc);
#pragma unroll
    for (int i = 0; i < 8; ++i) {
      const int m = i + half * 8, tok = t0 + m;
      if      (col < 8)   lds_dtraw[wv][m][col]              = (__bf16)acc[i];
      else if (col < 72)  Bs[(size_t)tok * N_ + (col - 8)]   = acc[i];
      else if (col < 136) Cs[(size_t)tok * N_ + (col - 72)]  = acc[i];
    }
  }
  __syncthreads();

  // ---- GEMM3: softplus(dt_raw @ W_dtr + b_dtr)  (K padded 8 -> 32)
  v16bf a3 = lda16(&lds_dtraw[wv][m16][0], 0, half);
  for (int nt = 0; nt < 8; ++nt) {
    const int col = nt * 16 + m16;
    const float bias = b_dtr[col];
    v8f acc;
#pragma unroll
    for (int i = 0; i < 8; ++i) acc[i] = bias;
    acc = wmma_bf16(a3, ldb16(wWdtrT + col * 32 + half * 16), acc);
#pragma unroll
    for (int i = 0; i < 8; ++i) {
      const int tok = t0 + i + half * 8;
      float v = acc[i];
      delta_out[(size_t)tok * C_ + col] = (v > 20.0f) ? v : log1pf(expf(v));
    }
  }
}

// ----------------- K2: depthwise 3x3 conv + z*silu(z), token-major layout
__global__ __launch_bounds__(256)
void k2_conv(const float* __restrict__ xcv, const float* __restrict__ K_conv,
             const float* __restrict__ b_conv, float* __restrict__ xc_tc)
{
  const int t  = blockIdx.x * blockDim.x + threadIdx.x;
  const int c  = t & 127, hw = (t >> 7) & 4095, b = t >> 19;   // c fastest
  const int h = hw >> 6, w = hw & 63;
  float acc = b_conv[c];
#pragma unroll
  for (int dy = 0; dy < 3; ++dy) {
    int hh = h + dy - 1; if ((unsigned)hh >= 64u) continue;
#pragma unroll
    for (int dx = 0; dx < 3; ++dx) {
      int ww = w + dx - 1; if ((unsigned)ww >= 64u) continue;
      acc += K_conv[c * 9 + dy * 3 + dx] *
             xcv[(size_t)((b << 12) + hh * 64 + ww) * C_ + c];
    }
  }
  float sig = 1.0f / (1.0f + expf(-acc));
  xc_tc[(size_t)((b << 12) + hw) * C_ + c] = acc * acc * sig;
}

// --------------------------------------- K3a: intra-chunk scan aggregates
__global__ __launch_bounds__(64)
void k3_intra(const float* __restrict__ delta, const float* __restrict__ u,
              const float* __restrict__ Bs, const float* __restrict__ Aneg,
              float* __restrict__ aA, float* __restrict__ aB)
{
  const int bi = blockIdx.x;                 // (b,chunk,c)
  const int c = bi & 127, chunk = (bi >> 7) & 63, b = bi >> 13;
  const int n = threadIdx.x;
  const float Acn = Aneg[c * N_ + n];
  float ca = 1.0f, cb = 0.0f;
  const int tbase = b * L_ + chunk * CHL;
  const float* pd = delta + (size_t)tbase * C_ + c;
  const float* pu = u     + (size_t)tbase * C_ + c;
  const float* pb = Bs    + (size_t)tbase * N_ + n;
  for (int i = 0; i < CHL; ++i) {
    float d = *pd, uu = *pu, bb = *pb;
    pd += C_; pu += C_; pb += N_;
    float da = expf(d * Acn);
    cb = da * cb + d * bb * uu;
    ca *= da;
  }
  const size_t idx = ((size_t)(b * C_ + c) * N_ + n) * NCH + chunk;
  aA[idx] = ca; aB[idx] = cb;
}

// --------------------- K3b: scan over chunk aggregates (aB[k] := entry state)
__global__ __launch_bounds__(256)
void k3_scan(const float* __restrict__ aA, float* __restrict__ aB)
{
  const int tid = blockIdx.x * blockDim.x + threadIdx.x;    // (b,c,n)
  const size_t base = (size_t)tid * NCH;
  float carry = 0.0f;
  for (int k = 0; k < NCH; ++k) {
    float ra = aA[base + k], rb = aB[base + k];
    aB[base + k] = carry;
    carry = ra * carry + rb;
  }
}

// ------- K3c: final pass; ybf = bf16( h.C + u*D + xc ), one wave per block
__global__ __launch_bounds__(32)
void k3_final(const float* __restrict__ delta, const float* __restrict__ u,
              const float* __restrict__ Bs, const float* __restrict__ Cs,
              const float* __restrict__ Aneg, const float* __restrict__ hinit,
              const float* __restrict__ Dp, const float* __restrict__ xc,
              __bf16* __restrict__ ybf)
{
  const int bi = blockIdx.x;                 // (b,chunk,c)
  const int c = bi & 127, chunk = (bi >> 7) & 63, b = bi >> 13;
  const int lane = threadIdx.x;
  const int n0 = lane, n1 = lane + 32;
  const size_t ib = (size_t)(b * C_ + c) * N_;
  float h0 = hinit[(ib + n0) * NCH + chunk];
  float h1 = hinit[(ib + n1) * NCH + chunk];
  const float A0 = Aneg[c * N_ + n0], A1 = Aneg[c * N_ + n1];
  const float Dc = Dp[c];
  const int tbase = b * L_ + chunk * CHL;
  const float* pd = delta + (size_t)tbase * C_ + c;
  const float* pu = u     + (size_t)tbase * C_ + c;
  const float* pb = Bs    + (size_t)tbase * N_;
  const float* pc = Cs    + (size_t)tbase * N_;
  const float* px = xc    + (size_t)tbase * C_ + c;
  __bf16* py = ybf + (size_t)tbase * C_ + c;
  for (int i = 0; i < CHL; ++i) {
    float d = *pd, uu = *pu;
    float b0 = pb[n0], b1 = pb[n1];
    float c0 = pc[n0], c1 = pc[n1];
    h0 = expf(d * A0) * h0 + d * b0 * uu;
    h1 = expf(d * A1) * h1 + d * b1 * uu;
    float part = h0 * c0 + h1 * c1;
#pragma unroll
    for (int off = 16; off > 0; off >>= 1) part += __shfl_xor(part, off, 32);
    if (lane == 0) *py = (__bf16)(part + uu * Dc + *px);
    pd += C_; pu += C_; pb += N_; pc += N_; px += C_; py += C_;
  }
}

// --------------------------- K4: out = ybf@W_out + x, transposed store
__global__ __launch_bounds__(128)
void k4_out(const float* __restrict__ x, const __bf16* __restrict__ ybf,
            const __bf16* __restrict__ wWoutT, float* __restrict__ out)
{
  const int lane = threadIdx.x & 31;
  const int wv   = threadIdx.x >> 5;
  const int t0   = (blockIdx.x * 4 + wv) * 16;
  const int m16  = lane & 15, half = lane >> 4;

  const __bf16* yrow = ybf + (size_t)(t0 + m16) * C_;
  v16bf afrag[4];
#pragma unroll
  for (int kt = 0; kt < 4; ++kt) afrag[kt] = lda16(yrow, kt, half);

  for (int nt = 0; nt < 8; ++nt) {
    const int col = nt * 16 + m16;
    const __bf16* bp = wWoutT + col * 128 + half * 16;
    v8f acc = {};
#pragma unroll
    for (int kt = 0; kt < 4; ++kt)
      acc = wmma_bf16(afrag[kt], ldb16(bp + kt * 32), acc);
#pragma unroll
    for (int i = 0; i < 8; ++i) {
      const int tok = t0 + i + half * 8;
      const int b = tok >> 12, hw = tok & 4095;
      const size_t idx = (size_t)(b * C_ + col) * L_ + hw;
      out[idx] = x[idx] + acc[i];
    }
  }
}

// ---------------------------------------------------------------------------
extern "C" void kernel_launch(void* const* d_in, const int* in_sizes, int n_in,
                              void* d_out, int out_size, void* d_ws, size_t ws_size,
                              hipStream_t stream)
{
  (void)in_sizes; (void)n_in; (void)out_size; (void)ws_size;
  const float* x      = (const float*)d_in[0];
  const float* W_in   = (const float*)d_in[1];
  const float* K_conv = (const float*)d_in[2];
  const float* b_conv = (const float*)d_in[3];
  const float* gamma  = (const float*)d_in[4];
  const float* beta   = (const float*)d_in[5];
  const float* W_dt   = (const float*)d_in[6];
  const float* b_dt   = (const float*)d_in[7];
  const float* W_dtr  = (const float*)d_in[8];
  const float* b_dtr  = (const float*)d_in[9];
  const float* A_log  = (const float*)d_in[10];
  const float* Dp     = (const float*)d_in[11];
  const float* W_out  = (const float*)d_in[12];
  float* out = (float*)d_out;

  char* ws = (char*)d_ws;                          // ~33.7 MB used
  __bf16* wWinT  = (__bf16*)(ws + 0);              // [256][128]
  __bf16* wWdtT  = (__bf16*)(ws + 65536);          // [144][128]
  __bf16* wWdtrT = (__bf16*)(ws + 102400);         // [128][32]
  __bf16* wWoutT = (__bf16*)(ws + 110592);         // [128][128]
  float*  Aneg   = (float*) (ws + 143360);         // [128][64]
  __bf16* xbf    = (__bf16*)(ws + 176128);         // [8192][128] bf16
  float*  u      = (float*) (ws + 2273280);        // [8192][128]
  float*  delta  = (float*) (ws + 6467584);        // [8192][128]
  float*  Bs     = (float*) (ws + 10661888);       // [8192][64]
  float*  Cs     = (float*) (ws + 12759040);       // [8192][64]
  float*  xcv    = (float*) (ws + 14856192);       // [8192][128] conv input
  float*  xc     = (float*) (ws + 19050496);       // [8192][128] conv output
  float*  aA     = (float*) (ws + 23244800);       // [(b,c,n)][64]
  float*  aB     = (float*) (ws + 27439104);       // aggregates -> entry states
  __bf16* ybf    = (__bf16*)(ws + 31633408);       // [8192][128] bf16

  k0_prep <<<312,   256, 0, stream>>>(W_in, W_dt, W_dtr, W_out, A_log,
                                      wWinT, wWdtT, wWdtrT, wWoutT, Aneg);
  k0b_xbf <<<4096,  256, 0, stream>>>(x, xbf);
  k1_proj <<<128,   128, 0, stream>>>(xbf, gamma, beta, b_dt, b_dtr,
                                      wWinT, wWdtT, wWdtrT,
                                      u, delta, Bs, Cs, xcv);
  k2_conv <<<4096,  256, 0, stream>>>(xcv, K_conv, b_conv, xc);
  k3_intra<<<16384,  64, 0, stream>>>(delta, u, Bs, Aneg, aA, aB);
  k3_scan <<<64,    256, 0, stream>>>(aA, aB);
  k3_final<<<16384,  32, 0, stream>>>(delta, u, Bs, Cs, Aneg, aB, Dp, xc, ybf);
  k4_out  <<<128,   128, 0, stream>>>(x, ybf, wWoutT, out);
}
